// DeepEdgeCongestionGNN_20693152432290
// MI455X (gfx1250) — matrix operand order repaired and checked by reading, not compiled, not run on hardware
//
#include <hip/hip_runtime.h>
#include <hip/hip_bf16.h>
#include <stdint.h>

#define IN_CH 8
#define HID 128
#define LAYERS 6
#define NODES_PER_GRAPH 30
#define NBRANCH 41
#define BN_EPS 1e-5f

typedef __attribute__((ext_vector_type(16))) __bf16 v16bf;
typedef __attribute__((ext_vector_type(8)))  __bf16 v8bf;
typedef __attribute__((ext_vector_type(8)))  float  v8f;

// ---- A-matrix fragment load: two 16B chunks (interleaved K layout per ISA 7.12.2)
__device__ __forceinline__ v16bf load_a_frag(const __bf16* __restrict__ row, int k0, int k1) {
    v8bf lo = *(const v8bf*)(row + k0);
    v8bf hi = *(const v8bf*)(row + k1);
    v16bf r;
#pragma unroll
    for (int i = 0; i < 8; ++i) { r[i] = lo[i]; r[i + 8] = hi[i]; }
    return r;
}
// ---- B-matrix fragment load: contiguous 16 K-values (32B) per lane
__device__ __forceinline__ v16bf load_b_frag(const __bf16* __restrict__ colrow, int k0) {
    return *(const v16bf*)(colrow + k0);
}

// ================= setup kernels =================
__global__ void deg_init_kernel(float* __restrict__ deg, int N) {
    int t = blockIdx.x * blockDim.x + threadIdx.x;
    if (t < N) deg[t] = 1.0f;   // self-loop
}
__global__ void deg_edge_kernel(const long long* __restrict__ ei, float* __restrict__ deg, long long E) {
    long long t = (long long)blockIdx.x * blockDim.x + threadIdx.x;
    if (t < E) atomicAdd(&deg[(int)ei[E + t]], 1.0f);
}
__global__ void dinv_kernel(float* __restrict__ deg, int N) {
    int t = blockIdx.x * blockDim.x + threadIdx.x;
    if (t < N) deg[t] = rsqrtf(deg[t]);       // deg >= 1 always (self-loop)
}
__global__ void norm_kernel(const long long* __restrict__ ei, const float* __restrict__ dinv,
                            float* __restrict__ nrm, long long E) {
    long long t = (long long)blockIdx.x * blockDim.x + threadIdx.x;
    if (t < E) nrm[t] = dinv[(int)ei[t]] * dinv[(int)ei[E + t]];
}

// transpose + bf16-convert weights: conv_W [L][K][N] -> wt [L][N][K]; mlp_W1 [256][128] -> w1t [128][256]
__global__ void prep_weights_kernel(const float* __restrict__ convW, const float* __restrict__ W1,
                                    __bf16* __restrict__ wt, __bf16* __restrict__ w1t) {
    int t = blockIdx.x * blockDim.x + threadIdx.x;
    const int CONVT = LAYERS * HID * HID;
    if (t < CONVT) {
        int l = t / (HID * HID);
        int rem = t - l * HID * HID;
        int n = rem / HID, k = rem - n * HID;
        wt[t] = (__bf16)convW[l * HID * HID + k * HID + n];
    } else if (t < CONVT + 2 * HID * HID) {
        int t2 = t - CONVT;
        int n = t2 / (2 * HID), k = t2 - n * (2 * HID);
        w1t[t2] = (__bf16)W1[k * HID + n];
    }
}

// encoder: h = x @ enc_W + enc_b   (8-deep dot, VALU is fine)
__global__ void encoder_kernel(const float* __restrict__ x, const float* __restrict__ W,
                               const float* __restrict__ b, float* __restrict__ h,
                               __bf16* __restrict__ hbf, int total) {
    int t = blockIdx.x * blockDim.x + threadIdx.x;
    if (t >= total) return;
    int n = t >> 7, c = t & (HID - 1);
    float s = b[c];
#pragma unroll
    for (int k = 0; k < IN_CH; ++k) s = fmaf(x[n * IN_CH + k], W[k * HID + c], s);
    h[t] = s;
    hbf[t] = (__bf16)s;
}

// ================= per-layer GEMM: m(bf16) = h(bf16) @ W^T(bf16), WMMA bf16->f32 =================
__global__ __launch_bounds__(256) void gcn_gemm_bf16(const __bf16* __restrict__ A,
                                                     const __bf16* __restrict__ Bt,
                                                     __bf16* __restrict__ out) {
    int lane = threadIdx.x & 31;
    int wv = threadIdx.x >> 5;
    int i = lane & 15;
    int row = blockIdx.x * 16 + i;       // node row for this lane's A fragment
    int col = wv * 16 + i;               // output column for this lane's B fragment
    const __bf16* arow = A + (size_t)row * HID;
    const __bf16* brow = Bt + (size_t)col * HID;
    v8f c = {};
#pragma unroll
    for (int s = 0; s < 4; ++s) {
        int ka = s * 32 + ((lane < 16) ? 0 : 8);
        int kb = s * 32 + ((lane < 16) ? 16 : 24);
        v16bf a = load_b_frag(arow, s * 32 + ((lane < 16) ? 0 : 16));
        v16bf bm = load_b_frag(brow, s * 32 + ((lane < 16) ? 0 : 16));
        c = __builtin_amdgcn_wmma_f32_16x16x32_bf16(false, a, false, bm, (short)0, c, false, false);
    }
    int rbase = blockIdx.x * 16 + ((lane < 16) ? 0 : 8);
#pragma unroll
    for (int r = 0; r < 8; ++r)
        out[(size_t)(rbase + r) * HID + col] = (__bf16)c[r];
}

// agg = (self-loop term) dinv[n]^2 * m[n]  -- also serves as accumulator init
__global__ void agg_init_kernel(const float* __restrict__ dinv, const __bf16* __restrict__ m,
                                float* __restrict__ agg, int total) {
    int t = blockIdx.x * blockDim.x + threadIdx.x;
    if (t >= total) return;
    float dn = dinv[t >> 7];
    agg[t] = dn * dn * (float)m[t];
}

// scatter: agg[dst] += norm[e] * m[src]; 4 features per thread; m gathers hit L2 (bf16 m fits in 192MB)
__global__ void edge_scatter_kernel(const long long* __restrict__ ei, const float* __restrict__ nrm,
                                    const __bf16* __restrict__ m, float* __restrict__ agg, long long E) {
    long long t = (long long)blockIdx.x * blockDim.x + threadIdx.x;
    if (t >= E * 32) return;
    int e = (int)(t >> 5);
    int c4 = ((int)t & 31) * 4;
    int src = (int)ei[e];
    int dst = (int)ei[E + e];
    float w = nrm[e];
    const __bf16* mp = m + (size_t)src * HID + c4;
    float* ap = agg + (size_t)dst * HID + c4;
#pragma unroll
    for (int i = 0; i < 4; ++i) atomicAdd(ap + i, w * (float)mp[i]);
}

// h = relu(bn(agg + bias)) + h ; also refresh bf16 copy for next layer's WMMA
__global__ void bn_relu_res_kernel(const float* __restrict__ agg, const float* __restrict__ b,
                                   const float* __restrict__ gm, const float* __restrict__ bt,
                                   const float* __restrict__ mu, const float* __restrict__ var,
                                   float* __restrict__ h, __bf16* __restrict__ hbf, int total) {
    int t = blockIdx.x * blockDim.x + threadIdx.x;
    if (t >= total) return;
    int c = t & (HID - 1);
    float v = agg[t] + b[c];
    v = (v - mu[c]) * rsqrtf(var[c] + BN_EPS) * gm[c] + bt[c];
    v = fmaxf(v, 0.0f);
    v += h[t];
    h[t] = v;
    hbf[t] = (__bf16)v;
}

// ================= fused readout: concat-gather + WMMA (K=256) + bias/ReLU + dot(W2) + b2 =======
__global__ __launch_bounds__(256) void readout_mlp_kernel(const __bf16* __restrict__ hbf,
                                                          const __bf16* __restrict__ w1t,
                                                          const float* __restrict__ b1,
                                                          const float* __restrict__ w2,
                                                          const float* __restrict__ b2,
                                                          const long long* __restrict__ bu,
                                                          const long long* __restrict__ bv,
                                                          float* __restrict__ out) {
    __shared__ float red[8][16];
    int lane = threadIdx.x & 31;
    int wv = threadIdx.x >> 5;
    int i = lane & 15;
    int r = blockIdx.x * 16 + i;                 // output row for this lane's A fragment
    int g = r / NBRANCH;
    int j = r - g * NBRANCH;
    size_t u = (size_t)g * NODES_PER_GRAPH + (size_t)bu[j];
    size_t v = (size_t)g * NODES_PER_GRAPH + (size_t)bv[j];
    const __bf16* rowu = hbf + u * HID;
    const __bf16* rowv = hbf + v * HID;
    int col = wv * 16 + i;
    const __bf16* brow = w1t + (size_t)col * (2 * HID);
    v8f c = {};
#pragma unroll
    for (int s = 0; s < 8; ++s) {
        const __bf16* arow = (s < 4) ? rowu : rowv;   // concat(h[u], h[v]) along K
        int ks = (s & 3) * 32;
        v16bf a = load_b_frag(arow, ks + ((lane < 16) ? 0 : 16));
        v16bf bm = load_b_frag(brow, s * 32 + ((lane < 16) ? 0 : 16));
        c = __builtin_amdgcn_wmma_f32_16x16x32_bf16(false, a, false, bm, (short)0, c, false, false);
    }
    float b1c = b1[col];
    float w2c = w2[col];
    float acc[8];
#pragma unroll
    for (int rr = 0; rr < 8; ++rr) acc[rr] = fmaxf(c[rr] + b1c, 0.0f) * w2c;
    // reduce over the 16 columns this wave owns (lanes 0-15 = rows 0-7, lanes 16-31 = rows 8-15)
#pragma unroll
    for (int mask = 1; mask < 16; mask <<= 1)
#pragma unroll
        for (int rr = 0; rr < 8; ++rr) acc[rr] += __shfl_xor(acc[rr], mask, 16);
    if (lane == 0)
#pragma unroll
        for (int rr = 0; rr < 8; ++rr) red[wv][rr] = acc[rr];
    if (lane == 16)
#pragma unroll
        for (int rr = 0; rr < 8; ++rr) red[wv][8 + rr] = acc[rr];
    __syncthreads();
    if (threadIdx.x < 16) {
        float s = 0.0f;
#pragma unroll
        for (int w = 0; w < 8; ++w) s += red[w][threadIdx.x];
        out[blockIdx.x * 16 + threadIdx.x] = s + b2[0];
    }
}

// ================================================================================================
extern "C" void kernel_launch(void* const* d_in, const int* in_sizes, int n_in,
                              void* d_out, int out_size, void* d_ws, size_t ws_size,
                              hipStream_t stream) {
    const float*     x      = (const float*)d_in[0];
    const long long* ei     = (const long long*)d_in[1];
    const float*     enc_W  = (const float*)d_in[3];
    const float*     enc_b  = (const float*)d_in[4];
    const float*     conv_W = (const float*)d_in[5];
    const float*     conv_b = (const float*)d_in[6];
    const float*     bn_g   = (const float*)d_in[7];
    const float*     bn_b   = (const float*)d_in[8];
    const float*     bn_m   = (const float*)d_in[9];
    const float*     bn_v   = (const float*)d_in[10];
    const float*     W1     = (const float*)d_in[11];
    const float*     b1     = (const float*)d_in[12];
    const float*     W2     = (const float*)d_in[13];
    const float*     b2     = (const float*)d_in[14];
    const long long* bu     = (const long long*)d_in[15];
    const long long* bv     = (const long long*)d_in[16];
    float* out = (float*)d_out;

    const int       N = in_sizes[0] / IN_CH;          // 491520
    const long long E = in_sizes[1] / 2;              // 1343488
    const int       G = N / NODES_PER_GRAPH;          // 16384
    const int       R = G * NBRANCH;                  // 671744 (multiple of 16)
    const int   total = N * HID;

    // ---- carve workspace (256B aligned slabs)
    char* base = (char*)d_ws;
    size_t off = 0;
    auto carve = [&](size_t bytes) { char* p = base + off; off += (bytes + 255) & ~(size_t)255; return p; };
    float*  dinv = (float*)carve((size_t)N * 4);
    float*  nrm  = (float*)carve((size_t)E * 4);
    float*  h    = (float*)carve((size_t)total * 4);
    float*  agg  = (float*)carve((size_t)total * 4);
    __bf16* hbf  = (__bf16*)carve((size_t)total * 2);
    __bf16* mbf  = (__bf16*)carve((size_t)total * 2);
    __bf16* wt   = (__bf16*)carve((size_t)LAYERS * HID * HID * 2);
    __bf16* w1t  = (__bf16*)carve((size_t)2 * HID * HID * 2);
    (void)ws_size; (void)n_in; (void)out_size;

    const int TB = 256;
    auto cdiv = [](long long a, long long b) { return (unsigned)((a + b - 1) / b); };

    // degrees / normalization
    deg_init_kernel<<<cdiv(N, TB), TB, 0, stream>>>(dinv, N);
    deg_edge_kernel<<<cdiv(E, TB), TB, 0, stream>>>(ei, dinv, E);
    dinv_kernel<<<cdiv(N, TB), TB, 0, stream>>>(dinv, N);
    norm_kernel<<<cdiv(E, TB), TB, 0, stream>>>(ei, dinv, nrm, E);

    // weight transpose + bf16 convert
    prep_weights_kernel<<<cdiv(LAYERS * HID * HID + 2 * HID * HID, TB), TB, 0, stream>>>(conv_W, W1, wt, w1t);

    // encoder
    encoder_kernel<<<cdiv(total, TB), TB, 0, stream>>>(x, enc_W, enc_b, h, hbf, total);

    // GCN layers
    for (int l = 0; l < LAYERS; ++l) {
        gcn_gemm_bf16<<<N / 16, TB, 0, stream>>>(hbf, wt + (size_t)l * HID * HID, mbf);
        agg_init_kernel<<<cdiv(total, TB), TB, 0, stream>>>(dinv, mbf, agg, total);
        edge_scatter_kernel<<<cdiv(E * 32, TB), TB, 0, stream>>>(ei, nrm, mbf, agg, E);
        bn_relu_res_kernel<<<cdiv(total, TB), TB, 0, stream>>>(
            agg, conv_b + l * HID, bn_g + l * HID, bn_b + l * HID,
            bn_m + l * HID, bn_v + l * HID, h, hbf, total);
    }

    // fused readout MLP
    readout_mlp_kernel<<<R / 16, TB, 0, stream>>>(hbf, w1t, b1, W2, b2, bu, bv, out);
}
